// OneLayerAttentionModel_78864189489552
// MI455X (gfx1250) — matrix operand order repaired and checked by reading, not compiled, not run on hardware
//
#include <hip/hip_runtime.h>

// ---------------------------------------------------------------------------
// One-layer attention LM forward for MI455X (gfx1250, wave32, WMMA bf16).
// Pipeline (all bf16 WMMA with fp32 accumulate):
//   h   = bf16(tok_emb[x] + pos_emb)                       [8192,1024]
//   Q/K/V = h @ W^T + b        (bf16 out)                  [8192,1024]
//   S   = (Q @ K^T) / 32       (fp32, batched)             [4,2048,2048]
//   P   = softmax(S)           (bf16)
//   O   = P @ V  (via V^T staged transpose, bf16 out)      [8192,1024]
//   out = O @ Wfc^T + bfc      (fp32)                      [8192,32000]
// GEMM uses double-buffered LDS with CDNA5 async global->LDS copies
// (ASYNCcnt) when the toolchain exposes them, overlapping HBM/L2 traffic
// with v_wmma_f32_16x16x32_bf16 issue.
// ---------------------------------------------------------------------------

typedef __attribute__((ext_vector_type(16))) __bf16 v16bf;
typedef __attribute__((ext_vector_type(8)))  __bf16 v8bf;
typedef __attribute__((ext_vector_type(8)))  float  v8f;

#define VOCAB 32000
#define DMODEL 1024
#define BATCH 4
#define SEQ 2048

union FragBF { v16bf v; v8bf h[2]; };

#if __has_builtin(__builtin_amdgcn_global_load_async_to_lds_b128) && \
    __has_builtin(__builtin_amdgcn_s_wait_asynccnt)
#define HAVE_ASYNC_LDS 1
// Builtin signature (from hipcc diagnostic): arg0 = v4i in AS1 (global,
// printed as __device__), arg1 = v4i in AS3 (LDS), then imm offset, imm cpol.
typedef int v4i __attribute__((vector_size(16)));
typedef v4i __attribute__((address_space(1))) v4i_global;
typedef v4i __attribute__((address_space(3))) v4i_lds;
#else
#define HAVE_ASYNC_LDS 0
#endif

// 16-byte global -> LDS copy: async on CDNA5, sync fallback otherwise.
__device__ __forceinline__ void cp16_g2l(const __bf16* g, __bf16* l) {
#if HAVE_ASYNC_LDS
  // Integer round-trip keeps the raw address bits: generic->AS1 is identity
  // for global memory; generic LDS pointers carry the 32-bit LDS offset in
  // the low bits, which is exactly what an AS3 pointer holds.
  __builtin_amdgcn_global_load_async_to_lds_b128(
      (v4i_global*)(uintptr_t)g, (v4i_lds*)(unsigned)(uintptr_t)l, 0, 0);
#else
  *(uint4*)l = *(const uint4*)g;
#endif
}

__device__ __forceinline__ void async_wait_all() {
#if HAVE_ASYNC_LDS
  __builtin_amdgcn_s_wait_asynccnt(0);
#endif
}

// --------------------------------------------------------------------------
// Embedding: h[bs,d] = bf16(tok_emb[x[bs], d] + pos_emb[s, d])
// --------------------------------------------------------------------------
__global__ __launch_bounds__(256) void embed_kernel(
    const int* __restrict__ x, const float* __restrict__ tok,
    const float* __restrict__ pos, __bf16* __restrict__ h) {
  size_t i = (size_t)blockIdx.x * 256 + threadIdx.x;  // over B*S*D
  int d = (int)(i % DMODEL);
  int bs = (int)(i / DMODEL);
  int s = bs % SEQ;
  float v = tok[(size_t)x[bs] * DMODEL + d] + pos[(size_t)s * DMODEL + d];
  h[i] = (__bf16)v;
}

// --------------------------------------------------------------------------
// fp32 -> bf16 conversion (weights)
// --------------------------------------------------------------------------
__global__ __launch_bounds__(256) void f2bf_kernel(
    const float* __restrict__ in, __bf16* __restrict__ out, size_t n) {
  size_t i = (size_t)blockIdx.x * 256 + threadIdx.x;
  if (i < n) out[i] = (__bf16)in[i];
}

// --------------------------------------------------------------------------
// Batched GEMM:  C[m,n] = scale * sum_k A[m,k] * B[n,k]  (+ bias[n])
// A: [M,K] bf16 row-major (K contiguous)
// B: [N,K] bf16 row-major (K contiguous)  == B-matrix columns are W rows
// C: fp32 or bf16, [M,N]
// Block tile 128x128, BK=32, 8 waves, each wave = 2x4 grid of 16x16 WMMA.
// Double-buffered LDS; tile k+1 streamed in (async) under tile k's WMMAs.
// --------------------------------------------------------------------------
template <bool OUT_BF16>
__global__ __launch_bounds__(256) void gemm_bf16_kernel(
    const __bf16* __restrict__ A, const __bf16* __restrict__ B, void* Cv,
    const float* __restrict__ bias, int M, int N, int K, float scale,
    long strideA, long strideB, long strideC) {
  constexpr int BM = 128, BN = 128, BK = 32;
  constexpr int LDT = BK + 8;  // LDS row stride (bf16), keeps 16B alignment
  __shared__ __bf16 sA[2][BM * LDT];
  __shared__ __bf16 sB[2][BN * LDT];

  const int t = threadIdx.x;
  const int lane = t & 31;
  const int wave = t >> 5;           // 0..7
  const int wm = wave & 3;           // 4 waves along M
  const int wn = wave >> 2;          // 2 waves along N
  const int half = lane >> 4;        // 0: lanes 0-15, 1: lanes 16-31
  const int lr = lane & 15;

  const int blockM = blockIdx.y * BM;
  const int blockN = blockIdx.x * BN;
  const __bf16* Ab = A + (size_t)blockIdx.z * strideA + (size_t)blockM * K;
  const __bf16* Bb = B + (size_t)blockIdx.z * strideB + (size_t)blockN * K;

  // Each thread stages two 16B chunks of A and two of B per K-tile:
  // chunk c (0..511) -> row c>>2, 16B quarter c&3.
  const int r0 = t >> 2, q0 = (t & 3) * 8;
  const int r1 = (t + 256) >> 2, q1 = ((t + 256) & 3) * 8;
  const size_t gA0 = (size_t)r0 * K + q0, gA1 = (size_t)r1 * K + q1;
  const int l0 = r0 * LDT + q0, l1 = r1 * LDT + q1;

  v8f acc[2][4];
#pragma unroll
  for (int mi = 0; mi < 2; ++mi)
#pragma unroll
    for (int ni = 0; ni < 4; ++ni) acc[mi][ni] = (v8f)0.0f;

  const int nk = K / BK;

  auto stage = [&](int kb, int buf) {
    const size_t ko = (size_t)kb * BK;
    cp16_g2l(Ab + ko + gA0, &sA[buf][l0]);
    cp16_g2l(Ab + ko + gA1, &sA[buf][l1]);
    cp16_g2l(Bb + ko + gA0, &sB[buf][l0]);
    cp16_g2l(Bb + ko + gA1, &sB[buf][l1]);
  };

  stage(0, 0);
  for (int kb = 0; kb < nk; ++kb) {
    const int buf = kb & 1;
    async_wait_all();    // this wave's copies into `buf` have landed
    __syncthreads();     // ... and every other wave's too
    if (kb + 1 < nk) stage(kb + 1, buf ^ 1);  // stream next tile under WMMAs

    // Fragment loads per ISA 16-bit A-layout (lanes 0-15: K 0..7,16..23;
    // lanes 16-31: K 8..15,24..31) and B-layout (same K order as A, for
    // column lr).
    FragBF af[2];
#pragma unroll
    for (int mi = 0; mi < 2; ++mi) {
      int row = wm * 32 + mi * 16 + lr;
      af[mi].h[0] = *(const v8bf*)&sA[buf][row * LDT + half * 8];
      af[mi].h[1] = *(const v8bf*)&sA[buf][row * LDT + 16 + half * 8];
    }
    FragBF bfr[4];
#pragma unroll
    for (int ni = 0; ni < 4; ++ni) {
      int col = wn * 64 + ni * 16 + lr;
      bfr[ni].h[0] = *(const v8bf*)&sB[buf][col * LDT + half * 8];
      bfr[ni].h[1] = *(const v8bf*)&sB[buf][col * LDT + 16 + half * 8];
    }
#pragma unroll
    for (int mi = 0; mi < 2; ++mi)
#pragma unroll
      for (int ni = 0; ni < 4; ++ni)
        acc[mi][ni] = __builtin_amdgcn_wmma_f32_16x16x32_bf16(
            false, af[mi].v, false, bfr[ni].v, (short)0, acc[mi][ni], false,
            false);
  }

  // Store C (D-matrix layout: element r -> row r + 8*half, col lr).
  __bf16* Cb = (__bf16*)Cv;
  float* Cf = (float*)Cv;
  size_t cbase = (size_t)blockIdx.z * strideC;
#pragma unroll
  for (int mi = 0; mi < 2; ++mi) {
#pragma unroll
    for (int ni = 0; ni < 4; ++ni) {
      int col = blockN + wn * 64 + ni * 16 + lr;
      float bv = bias ? bias[col] : 0.0f;
#pragma unroll
      for (int r = 0; r < 8; ++r) {
        int row = blockM + wm * 32 + mi * 16 + half * 8 + r;
        float v = acc[mi][ni][r] * scale + bv;
        size_t idx = cbase + (size_t)row * N + col;
        if constexpr (OUT_BF16)
          Cb[idx] = (__bf16)v;
        else
          Cf[idx] = v;
      }
    }
  }
}

// --------------------------------------------------------------------------
// Row softmax: one 256-thread block per row of length SEQ; bf16 output.
// --------------------------------------------------------------------------
__global__ __launch_bounds__(256) void softmax_kernel(
    const float* __restrict__ scores, __bf16* __restrict__ attn) {
  const size_t base = (size_t)blockIdx.x * SEQ;
  const int t = threadIdx.x;
  __shared__ float red[256];

  float v[SEQ / 256];
  float m = -3.4e38f;
#pragma unroll
  for (int i = 0; i < SEQ / 256; ++i) {
    v[i] = scores[base + t + 256 * i];
    m = fmaxf(m, v[i]);
  }
  red[t] = m;
  __syncthreads();
  for (int s = 128; s > 0; s >>= 1) {
    if (t < s) red[t] = fmaxf(red[t], red[t + s]);
    __syncthreads();
  }
  m = red[0];
  __syncthreads();

  float sum = 0.0f;
#pragma unroll
  for (int i = 0; i < SEQ / 256; ++i) {
    v[i] = __expf(v[i] - m);
    sum += v[i];
  }
  red[t] = sum;
  __syncthreads();
  for (int s = 128; s > 0; s >>= 1) {
    if (t < s) red[t] += red[t + s];
    __syncthreads();
  }
  float inv = 1.0f / red[0];
#pragma unroll
  for (int i = 0; i < SEQ / 256; ++i)
    attn[base + t + 256 * i] = (__bf16)(v[i] * inv);
}

// --------------------------------------------------------------------------
// Batched 32x32 LDS transpose (bf16): out[b][c][r] = in[b][r][c]
// --------------------------------------------------------------------------
__global__ __launch_bounds__(256) void transpose_kernel(
    const __bf16* __restrict__ in, __bf16* __restrict__ out, int rows,
    int cols, long strideIn, long strideOut) {
  __shared__ __bf16 tile[32][33];
  const int tx = threadIdx.x & 31;
  const int ty = threadIdx.x >> 5;  // 0..7
  const int r0 = blockIdx.y * 32;
  const int c0 = blockIdx.x * 32;
  in += (size_t)blockIdx.z * strideIn;
  out += (size_t)blockIdx.z * strideOut;
#pragma unroll
  for (int i = 0; i < 32; i += 8)
    tile[ty + i][tx] = in[(size_t)(r0 + ty + i) * cols + c0 + tx];
  __syncthreads();
#pragma unroll
  for (int i = 0; i < 32; i += 8)
    out[(size_t)(c0 + ty + i) * rows + r0 + tx] = tile[tx][ty + i];
}

// --------------------------------------------------------------------------
extern "C" void kernel_launch(void* const* d_in, const int* in_sizes, int n_in,
                              void* d_out, int out_size, void* d_ws,
                              size_t ws_size, hipStream_t stream) {
  const int* x = (const int*)d_in[0];
  const float* tok_emb = (const float*)d_in[1];
  const float* pos_emb = (const float*)d_in[2];
  const float* Wq = (const float*)d_in[3];
  const float* bq = (const float*)d_in[4];
  const float* Wk = (const float*)d_in[5];
  const float* bk = (const float*)d_in[6];
  const float* Wv = (const float*)d_in[7];
  const float* bv = (const float*)d_in[8];
  const float* Wfc = (const float*)d_in[9];
  const float* bfc = (const float*)d_in[10];
  float* out = (float*)d_out;

  const size_t BS = (size_t)BATCH * SEQ;      // 8192
  const size_t BSD = BS * DMODEL;             // 8.4M elems
  const size_t DD = (size_t)DMODEL * DMODEL;  // 1M
  const size_t VD = (size_t)VOCAB * DMODEL;   // 32.7M
  const size_t SS = (size_t)SEQ * SEQ;        // 4.2M per batch

  // Workspace layout (256B aligned slices).
  char* ws = (char*)d_ws;
  size_t off = 0;
  auto take = [&](size_t bytes) {
    char* p = ws + off;
    off = (off + bytes + 255) & ~(size_t)255;
    return p;
  };
  __bf16* hbf = (__bf16*)take(BSD * 2);
  __bf16* Wq_b = (__bf16*)take(DD * 2);
  __bf16* Wk_b = (__bf16*)take(DD * 2);
  __bf16* Wv_b = (__bf16*)take(DD * 2);
  __bf16* Wfc_b = (__bf16*)take(VD * 2);
  __bf16* Qb = (__bf16*)take(BSD * 2);
  __bf16* Kb = (__bf16*)take(BSD * 2);
  __bf16* Vb = (__bf16*)take(BSD * 2);
  float* scores = (float*)take(BATCH * SS * 4);
  __bf16* attn = (__bf16*)take(BATCH * SS * 2);
  __bf16* Vt = (__bf16*)take(BSD * 2);
  __bf16* Ob = (__bf16*)take(BSD * 2);

  // 1. Embedding -> bf16 activations
  embed_kernel<<<BSD / 256, 256, 0, stream>>>(x, tok_emb, pos_emb, hbf);

  // 2. Weights -> bf16
  f2bf_kernel<<<(DD + 255) / 256, 256, 0, stream>>>(Wq, Wq_b, DD);
  f2bf_kernel<<<(DD + 255) / 256, 256, 0, stream>>>(Wk, Wk_b, DD);
  f2bf_kernel<<<(DD + 255) / 256, 256, 0, stream>>>(Wv, Wv_b, DD);
  f2bf_kernel<<<(VD + 255) / 256, 256, 0, stream>>>(Wfc, Wfc_b, VD);

  // 3. Q, K, V projections: [8192,1024] = h @ W^T + b (bf16 out)
  dim3 gQKV(DMODEL / 128, BS / 128, 1);
  gemm_bf16_kernel<true><<<gQKV, 256, 0, stream>>>(
      hbf, Wq_b, Qb, bq, (int)BS, DMODEL, DMODEL, 1.0f, 0, 0, 0);
  gemm_bf16_kernel<true><<<gQKV, 256, 0, stream>>>(
      hbf, Wk_b, Kb, bk, (int)BS, DMODEL, DMODEL, 1.0f, 0, 0, 0);
  gemm_bf16_kernel<true><<<gQKV, 256, 0, stream>>>(
      hbf, Wv_b, Vb, bv, (int)BS, DMODEL, DMODEL, 1.0f, 0, 0, 0);

  // 4. scores = Q @ K^T / sqrt(D)   (batched, fp32 out)
  dim3 gScore(SEQ / 128, SEQ / 128, BATCH);
  gemm_bf16_kernel<false><<<gScore, 256, 0, stream>>>(
      Qb, Kb, scores, nullptr, SEQ, SEQ, DMODEL, 0.03125f, (long)(SEQ * DMODEL),
      (long)(SEQ * DMODEL), (long)SS);

  // 5. softmax rows -> bf16 probabilities
  softmax_kernel<<<(unsigned)BS, 256, 0, stream>>>(scores, attn);

  // 6. V^T per batch so P@V's B-operand is K-contiguous
  dim3 gT(DMODEL / 32, SEQ / 32, BATCH);
  transpose_kernel<<<gT, 256, 0, stream>>>(Vb, Vt, SEQ, DMODEL,
                                           (long)(SEQ * DMODEL),
                                           (long)(SEQ * DMODEL));

  // 7. O = P @ V  (batched, bf16 out)
  dim3 gPV(DMODEL / 128, SEQ / 128, BATCH);
  gemm_bf16_kernel<true><<<gPV, 256, 0, stream>>>(
      attn, Vt, Ob, nullptr, SEQ, DMODEL, SEQ, 1.0f, (long)SS,
      (long)(SEQ * DMODEL), (long)(SEQ * DMODEL));

  // 8. logits = O @ Wfc^T + bfc  (fp32 out, the 537 GF GEMM)
  dim3 gFC(VOCAB / 128, BS / 128, 1);
  gemm_bf16_kernel<false><<<gFC, 256, 0, stream>>>(
      Ob, Wfc_b, out, bfc, (int)BS, VOCAB, DMODEL, 1.0f, 0, 0, 0);
}
